// sLSTM_31044023616495
// MI455X (gfx1250) — compile-verified, matrix operand order checked by reading, not run on hardware
//
#include <hip/hip_runtime.h>

#define B_   16
#define S_   2048
#define E_   512
#define NH_  4
#define DH_  128
#define H_   512
#define G4H_ 2048
#define PADR 136   // padded bf16 row stride (elems) for LDS tiles: 272B -> conflict-free

typedef __bf16 bf16;
typedef __bf16 v16bf __attribute__((ext_vector_type(16)));
typedef __bf16 v8bf  __attribute__((ext_vector_type(8)));
typedef float  v8f   __attribute__((ext_vector_type(8)));
typedef unsigned int v4u __attribute__((ext_vector_type(4)));
typedef int v8i __attribute__((ext_vector_type(8)));
typedef int v4i __attribute__((ext_vector_type(4)));

#if defined(__has_builtin)
#if __has_builtin(__builtin_amdgcn_tensor_load_to_lds)
#define HAVE_TDM 1
#endif
#endif
#ifndef HAVE_TDM
#define HAVE_TDM 0
#endif

// Load a 16x32 bf16 WMMA A-fragment (rows = lane&15) or, identically, a 32x16
// B-fragment from an out-major (row = N, col = K) source. Matches the CDNA5
// 16-bit A/B VGPR layout: lanes 0-15 hold K 0..7 / 16..23, lanes 16-31 hold
// K 8..15 / 24..31 of row (lane&15). Two 16-byte loads per lane.
__device__ __forceinline__ v16bf load_frag(const bf16* base, int stride) {
  const int lane = (int)(threadIdx.x & 31u);
  const bf16* p = base + (lane & 15) * stride + ((lane >> 4) << 3);
  v8bf lo = *(const v8bf*)(p);
  v8bf hi = *(const v8bf*)(p + 16);
  v16bf r;
#pragma unroll
  for (int i = 0; i < 8; ++i) { r[i] = lo[i]; r[i + 8] = hi[i]; }
  return r;
}

// ---------------- prep kernels ----------------

__global__ void cvt_silu_kernel(const float* __restrict__ x, bf16* __restrict__ xbf,
                                bf16* __restrict__ xsbf, int n) {
  int i = blockIdx.x * 256 + threadIdx.x;
  if (i < n) {
    float v = x[i];
    xbf[i]  = (bf16)v;
    xsbf[i] = (bf16)(v * (1.f / (1.f + __expf(-v))));
  }
}

__global__ void cvt_w_kernel(const float* __restrict__ w, bf16* __restrict__ d, int n) {
  int i = blockIdx.x * 256 + threadIdx.x;
  if (i < n) d[i] = (bf16)w[i];
}

// ReffT[h][g][a] = rkernel[h][(a&31)*4+(g>>7)][(g>>5)&3][(g&31)*4+(a>>5)]
// (derived from the reference's transpose/reshape chain); out-major for B-frags.
__global__ void reff_kernel(const float* __restrict__ rk, bf16* __restrict__ rt) {
  int i = blockIdx.x * 256 + threadIdx.x;
  if (i >= NH_ * H_ * DH_) return;
  const int a = i & 127;
  const int g = (i >> 7) & 511;
  const int h = i >> 16;
  const int d1 = (a & 31) * 4 + (g >> 7);
  const int gk = (g >> 5) & 3;
  const int d3 = (g & 31) * 4 + (a >> 5);
  rt[i] = (bf16)rk[((h * DH_ + d1) * 4 + gk) * DH_ + d3];
}

// ---------------- phase 1: wx = proj(x) ----------------
// grid (S, NH), 256 threads. M-tile rows = batch (all 16 b at one t).
// wx layout: [t][b][gate*512 + hh*128 + dout]  (fp32)

__global__ void __launch_bounds__(256) gemm_wx_kernel(
    const bf16* __restrict__ xbf, const bf16* __restrict__ xsbf,
    const bf16* __restrict__ wbf, float* __restrict__ wx) {
  const int t = blockIdx.x, hh = blockIdx.y;
  const int wid = threadIdx.x >> 5, lane = threadIdx.x & 31;
  const int gate = wid >> 1;                      // gates 0,1 use silu(x)
  const bf16* xa = (gate < 2) ? xsbf : xbf;
  v16bf A[4];
#pragma unroll
  for (int kb = 0; kb < 4; ++kb)
    A[kb] = load_frag(xa + t * E_ + hh * DH_ + kb * 32, S_ * E_);
#pragma unroll
  for (int i = 0; i < 4; ++i) {
    const int dout = ((((wid & 1) << 2) + i) << 4);
    v8f c = {0.f, 0.f, 0.f, 0.f, 0.f, 0.f, 0.f, 0.f};
#pragma unroll
    for (int kb = 0; kb < 4; ++kb) {
      v16bf b = load_frag(wbf + ((gate * NH_ + hh) * DH_ + dout) * DH_ + kb * 32, DH_);
      c = __builtin_amdgcn_wmma_f32_16x16x32_bf16(false, A[kb], false, b,
                                                  (short)0, c, false, false);
    }
    const int col = gate * H_ + hh * DH_ + dout + (lane & 15);
    const int b0 = (lane >> 4) << 3;
    float* op = wx + ((size_t)t * B_ + b0) * G4H_ + col;
#pragma unroll
    for (int r = 0; r < 8; ++r)
      __builtin_nontemporal_store(c[r], op + (size_t)r * G4H_);
  }
}

// ---------------- phase 2+3: scan + fused group-norm ----------------

#if HAVE_TDM
typedef const __attribute__((address_space(3))) char* lds_cptr;
__device__ __forceinline__ unsigned lds_addr_of(const void* p) {
  return (unsigned)(unsigned long long)(lds_cptr)p;   // LDS byte offset
}
// TDM: 2D tile, 64 rows (k = 4*b+gate, stride 512 fp32) x 128 fp32 -> 32KB LDS
__device__ __forceinline__ void tdm_load_wx(const float* wx, int t, int hh,
                                            unsigned lds_off) {
  unsigned long long ga =
      reinterpret_cast<unsigned long long>(wx + (size_t)t * (B_ * G4H_) + hh * DH_);
  v4u g0 = {1u,                                   // count=1, user descriptor
            lds_off,                              // lds_addr
            (unsigned)ga,                         // global_addr[31:0]
            (unsigned)((ga >> 32) & 0x1FFFFFFull) | 0x80000000u}; // addr[56:32]|type=2
  v8i g1 = {(int)(2u << 16),     // data_size = 4B
            (int)(512u << 16),   // tensor_dim0 = 512
            (int)(64u << 16),    // tensor_dim1 = 64
            (int)(128u << 16),   // tile_dim0 = 128
            64,                  // tile_dim1 = 64, tile_dim2 = 0
            512,                 // tensor_dim0_stride = 512
            0, 0};
  v4i z4 = {0, 0, 0, 0};
  v8i z8 = {0, 0, 0, 0, 0, 0, 0, 0};
  __builtin_amdgcn_tensor_load_to_lds(g0, g1, z4, z4, z8, 0);
}
#endif

// LDS byte map (dynamic shared, 241920 B total):
//   [0)        ReffT padded    512*136*2 = 139264
//   [139264)   hstate bf16      16*136*2 =   4352
//   [143616)   raw fp32       4*16*128*4 =  32768
//   [176384)   wx double buf  2*8192*4   =  65536
__global__ void __launch_bounds__(256) scan_kernel(
    const float* __restrict__ wx, const bf16* __restrict__ reffT,
    const float* __restrict__ bias, float* __restrict__ out) {
  extern __shared__ char smem[];
  bf16*  rt  = (bf16*)(smem);
  bf16*  hs  = (bf16*)(smem + 139264);
  float* raw = (float*)(smem + 143616);
  float* wxs = (float*)(smem + 176384);
  const int hh = blockIdx.x;
  const int tid = threadIdx.x, wid = tid >> 5, lane = tid & 31;
  const int gate = wid >> 1;

  // stage this head's ReffT into LDS (padded rows), zero hidden state
  {
    const bf16* src = reffT + (size_t)hh * H_ * DH_;
#pragma unroll
    for (int rr = 0; rr < 2; ++rr) {
      const int r = tid * 2 + rr;
      const uint4* s = (const uint4*)(src + r * DH_);
      uint4* d = (uint4*)(rt + r * PADR);
#pragma unroll
      for (int q = 0; q < 16; ++q) d[q] = s[q];
    }
  }
  for (int i = tid; i < B_ * PADR; i += 256) hs[i] = (bf16)0.f;

  float biasv[4];
#pragma unroll
  for (int i = 0; i < 4; ++i) {
    const int dout = ((((wid & 1) << 2) + i) << 4);
    biasv[i] = bias[gate * H_ + hh * DH_ + dout + (lane & 15)];
  }

  const int sb  = tid >> 4;          // this thread's batch row
  const int sd0 = (tid & 15) << 3;   // 8 consecutive channels
  float cst[8], mst[8], hst[8];
#pragma unroll
  for (int j = 0; j < 8; ++j) { cst[j] = 0.f; mst[j] = 0.f; hst[j] = 0.f; }

#if HAVE_TDM
  if (tid < 32) tdm_load_wx(wx, 0, hh, lds_addr_of(wxs));
#endif

  for (int t = 0; t < S_; ++t) {
#if HAVE_TDM
    if (tid < 32) __builtin_amdgcn_s_wait_tensorcnt(0);
    __syncthreads();   // wx[t&1] + hstate(t) visible; buf[(t+1)&1] free
    if (tid < 32 && t + 1 < S_)
      tdm_load_wx(wx, t + 1, hh, lds_addr_of(wxs + ((t + 1) & 1) * 8192));
    const float* wxt = wxs + (t & 1) * 8192;
#else
    for (int i = tid; i < 2048; i += 256) {
      const int r = i >> 5, q = i & 31;
      ((uint4*)wxs)[i] =
          ((const uint4*)(wx + (size_t)t * (B_ * G4H_) + hh * DH_ + r * H_))[q];
    }
    __syncthreads();
    const float* wxt = wxs;
#endif
    // ---- recurrent matmul: raw = h @ ReffT + bias ----
    v16bf A[4];
#pragma unroll
    for (int kb = 0; kb < 4; ++kb) A[kb] = load_frag(hs + kb * 32, PADR);
#pragma unroll
    for (int i = 0; i < 4; ++i) {
      const int dout = ((((wid & 1) << 2) + i) << 4);
      const float bv = biasv[i];
      v8f c = {bv, bv, bv, bv, bv, bv, bv, bv};
#pragma unroll
      for (int kb = 0; kb < 4; ++kb) {
        v16bf b = load_frag(rt + (gate * DH_ + dout) * PADR + kb * 32, PADR);
        c = __builtin_amdgcn_wmma_f32_16x16x32_bf16(false, A[kb], false, b,
                                                    (short)0, c, false, false);
      }
      const int col = dout + (lane & 15);
      const int b0 = (lane >> 4) << 3;
#pragma unroll
      for (int r = 0; r < 8; ++r)
        raw[(gate * B_ + b0 + r) * DH_ + col] = c[r];
    }
    __syncthreads();   // raw ready; A-frag reads of hs done
    // ---- gates + state update + fused group-norm ----
    float hloc[8], sum = 0.f, sumsq = 0.f;
#pragma unroll
    for (int j = 0; j < 8; ++j) {
      const int d = sd0 + j;
      const float ir  = raw[(0 * B_ + sb) * DH_ + d] + wxt[(4 * sb + 0) * DH_ + d];
      const float fr  = raw[(1 * B_ + sb) * DH_ + d] + wxt[(4 * sb + 1) * DH_ + d];
      const float zr  = raw[(2 * B_ + sb) * DH_ + d] + wxt[(4 * sb + 2) * DH_ + d];
      const float orr = raw[(3 * B_ + sb) * DH_ + d] + wxt[(4 * sb + 3) * DH_ + d];
      const float lsf  = fminf(fr, 0.f) - log1pf(__expf(-fabsf(fr)));
      const float lfm  = mst[j] + lsf;
      const float mnew = (t == 0) ? ir : fmaxf(ir, lfm);
      const float og = 1.f / (1.f + __expf(-orr));
      const float ig = __expf(ir - mnew);
      const float fg = __expf(lfm - mnew);
      const float cn = fg * cst[j] + ig * tanhf(zr);
      const float nn = fg * hst[j] + ig;      // note: fgate*h per reference
      const float hn = og * cn / nn;
      cst[j] = cn; mst[j] = mnew; hst[j] = hn; hloc[j] = hn;
      sum += hn; sumsq += hn * hn;
      hs[sb * PADR + d] = (bf16)hn;
    }
#pragma unroll
    for (int m = 1; m < 16; m <<= 1) {        // reduce over the 16 lanes of sb
      sum   += __shfl_xor(sum, m, 32);
      sumsq += __shfl_xor(sumsq, m, 32);
    }
    const float mu = sum * (1.f / 128.f);
    const float rs = rsqrtf(sumsq * (1.f / 128.f) - mu * mu + 1e-5f);
    float* op = out + ((size_t)sb * S_ + t) * H_ + hh * DH_ + sd0;
#pragma unroll
    for (int j = 0; j < 8; ++j)
      __builtin_nontemporal_store((hloc[j] - mu) * rs, op + j);
#if !HAVE_TDM
    __syncthreads();   // wxs reuse guard for the fallback copy
#endif
  }
}

// ---------------- launcher ----------------

extern "C" void kernel_launch(void* const* d_in, const int* in_sizes, int n_in,
                              void* d_out, int out_size, void* d_ws, size_t ws_size,
                              hipStream_t stream) {
  (void)in_sizes; (void)n_in; (void)out_size; (void)ws_size;
  const float* x  = (const float*)d_in[0];
  const float* wf = (const float*)d_in[1];
  const float* wi = (const float*)d_in[2];
  const float* wz = (const float*)d_in[3];
  const float* wo = (const float*)d_in[4];
  const float* rk = (const float*)d_in[5];
  const float* bs = (const float*)d_in[6];
  float* out = (float*)d_out;
  char* ws = (char*)d_ws;
  bf16*  xbf   = (bf16*)(ws + 0);          // 32 MB
  bf16*  xsbf  = (bf16*)(ws + 33554432);   // 32 MB
  bf16*  wbf   = (bf16*)(ws + 67108864);   // 512 KB
  bf16*  reffT = (bf16*)(ws + 67633152);   // 512 KB
  float* wx    = (float*)(ws + 68157440);  // 256 MB

  const int n = B_ * S_ * E_;
  cvt_silu_kernel<<<n / 256, 256, 0, stream>>>(x, xbf, xsbf, n);
  const int nw = NH_ * DH_ * DH_;
  cvt_w_kernel<<<nw / 256, 256, 0, stream>>>(wf, wbf + 0 * nw, nw);  // gate0 = i
  cvt_w_kernel<<<nw / 256, 256, 0, stream>>>(wi, wbf + 1 * nw, nw);  // gate1 = f
  cvt_w_kernel<<<nw / 256, 256, 0, stream>>>(wz, wbf + 2 * nw, nw);  // gate2 = z
  cvt_w_kernel<<<nw / 256, 256, 0, stream>>>(wo, wbf + 3 * nw, nw);  // gate3 = o
  reff_kernel<<<(NH_ * H_ * DH_) / 256, 256, 0, stream>>>(rk, reffT);
  gemm_wx_kernel<<<dim3(S_, NH_), 256, 0, stream>>>(xbf, xsbf, wbf, wx);
  scan_kernel<<<NH_, 256, 241920, stream>>>(wx, reffT, bs, out);
}
